// Codebook_80900003987995
// MI455X (gfx1250) — matrix operand order repaired and checked by reading, not compiled, not run on hardware
//
#include <hip/hip_runtime.h>
#include <hip/hip_bf16.h>
#include <cstdint>

// ---------------------------------------------------------------------------
// VQ-VAE codebook quantization for MI455X (gfx1250, wave32, WMMA).
//   z:   [B,C,H,W] f32   (C=64, H*W=4096)
//   emb: [K,C]     f32   (K=1024)
// Outputs (concatenated in d_out as float):
//   z_q (transposed gather)  [B,C,H,W]
//   indices (as float)       [B,H,W]
//   loss (scalar)            [1]
// Core: scores = ||e||^2 - 2*z@e^T via 3xBF16-split WMMA (fp32-class accuracy)
// v3: widen VGPR budget (launch_bounds + waves_per_eu) to kill the scratch
//     spills seen in v2's inner loop; occupancy is LDS-bound so VGPRs are free.
// ---------------------------------------------------------------------------

typedef __bf16 v16bf __attribute__((ext_vector_type(16)));
typedef float  v8f   __attribute__((ext_vector_type(8)));

#define C_DIM 64
#define HW    4096
#define ROWS_PER_BLOCK 256   // 8 waves * 32 rows

union Frag {
    unsigned short u[16];
    uint4          q[2];
    v16bf          v;
};

__device__ __forceinline__ unsigned short f2bf(float f) {
    unsigned u = __float_as_uint(f);
    unsigned r = u + 0x7FFFu + ((u >> 16) & 1u);   // round-to-nearest-even
    return (unsigned short)(r >> 16);
}
__device__ __forceinline__ float bf2f(unsigned short h) {
    return __uint_as_float(((unsigned)h) << 16);
}

// ---------------------------------------------------------------------------
// Prep: swizzle embedding into per-lane bf16 B-fragments (hi/lo split) + norms.
// ws layout: [K/16 tiles][4 frags][32 lanes][32 bytes]  then  K floats (norms)
// B layout used here (contiguous k, same as A): lane l -> column N=l%16; element e -> K = e + (l>=16)*16
// frag 0 = hi chunk0 (c 0..31), 1 = hi chunk1, 2 = lo chunk0, 3 = lo chunk1
// ---------------------------------------------------------------------------
__global__ void vq_prep(const float* __restrict__ emb,
                        unsigned char* __restrict__ ws, int K) {
    const int t    = blockIdx.x;      // k-tile
    const int lane = threadIdx.x;     // 0..31
    const int code = t * 16 + (lane & 15);

    Frag hi0, hi1, lo0, lo1;
#pragma unroll
    for (int e = 0; e < 16; ++e) {
        const int c0 = ((lane < 16) ? e : e + 16);
        float f0 = emb[code * C_DIM + c0];
        unsigned short h0 = f2bf(f0);
        hi0.u[e] = h0;
        lo0.u[e] = f2bf(f0 - bf2f(h0));

        const int c1 = c0 + 32;
        float f1 = emb[code * C_DIM + c1];
        unsigned short h1 = f2bf(f1);
        hi1.u[e] = h1;
        lo1.u[e] = f2bf(f1 - bf2f(h1));
    }
    unsigned char* base = ws + (size_t)t * 4096 + (size_t)lane * 32;
    ((uint4*)(base + 0 * 1024))[0] = hi0.q[0];
    ((uint4*)(base + 0 * 1024))[1] = hi0.q[1];
    ((uint4*)(base + 1 * 1024))[0] = hi1.q[0];
    ((uint4*)(base + 1 * 1024))[1] = hi1.q[1];
    ((uint4*)(base + 2 * 1024))[0] = lo0.q[0];
    ((uint4*)(base + 2 * 1024))[1] = lo0.q[1];
    ((uint4*)(base + 3 * 1024))[0] = lo1.q[0];
    ((uint4*)(base + 3 * 1024))[1] = lo1.q[1];

    if (lane < 16) {
        float s = 0.f;
#pragma unroll
        for (int c = 0; c < C_DIM; ++c) {
            float v = emb[code * C_DIM + c];
            s += v * v;
        }
        float* norms = (float*)(ws + (size_t)(K / 16) * 4096);
        norms[code] = s;
    }
}

__global__ void vq_zero_loss(float* loss) { *loss = 0.f; }

// ---------------------------------------------------------------------------
// Main kernel: 256 threads = 8 waves; each wave owns 32 rows (2 groups of 16).
// All K codes (swizzled bf16 frags + norms) staged in LDS (~300 KB of 320 KB).
// Occupancy is LDS-bound (1 WG/WGP -> 2 waves/SIMD), so request a large VGPR
// budget to keep all fragments + accumulators + argmin state in registers.
// ---------------------------------------------------------------------------
__global__ void
__launch_bounds__(256, 1)
__attribute__((amdgpu_waves_per_eu(1, 2)))
vq_main(const float* __restrict__ z,
        const float* __restrict__ emb,
        const unsigned char* __restrict__ ws,
        float* __restrict__ out_zq,
        float* __restrict__ out_idx,
        float* __restrict__ out_loss,
        int K, float lossScale) {
    extern __shared__ char smem[];
    const int tid  = threadIdx.x;
    const int wave = tid >> 5;
    const int lane = tid & 31;
    const int nTiles = K / 16;

    const size_t bBytes = (size_t)nTiles * 4096;
    char*  sB    = smem;
    float* sNorm = (float*)(smem + bBytes);
    char*  p     = smem + bBytes + (size_t)K * 4;
    float* sMinV = (float*)p;  p += ROWS_PER_BLOCK * 16 * 4;
    int*   sMinI = (int*)p;    p += ROWS_PER_BLOCK * 16 * 4;
    float* sZn   = (float*)p;  p += ROWS_PER_BLOCK * 4;
    int*   sIdx  = (int*)p;    p += ROWS_PER_BLOCK * 4;
    float* sLoss = (float*)p;

    // ---- stage swizzled codebook (frags + norms) into LDS ----
    {
        const uint4* src = (const uint4*)ws;
        uint4* dst = (uint4*)smem;
        const int nVec = (int)((bBytes + (size_t)K * 4) / 16);
        for (int i = tid; i < nVec; i += blockDim.x) dst[i] = src[i];
    }
    if (tid == 0) sLoss[0] = 0.f;

    // ---- build A fragments (z rows, bf16 hi/lo split) while LDS fills ----
    // A layout used here (contiguous k, same as B): lane l -> row M=l%16;
    // element e -> K = e + (l>=16)*16
    const int m     = lane & 15;
    const int kbase = (lane < 16) ? 0 : 16;

    Frag Ahi[2][2], Alo[2][2];           // [row group][K chunk]
    float zn[2];
#pragma unroll
    for (int g = 0; g < 2; ++g) {
        const int row  = blockIdx.x * ROWS_PER_BLOCK + wave * 32 + g * 16 + m;
        const int bImg = row >> 12;      // H*W = 4096
        const int hw   = row & 4095;
        const size_t zRowBase = (size_t)bImg * C_DIM * HW + (size_t)hw;
        float acc = 0.f;
#pragma unroll
        for (int ch = 0; ch < 2; ++ch) {
#pragma unroll
            for (int e = 0; e < 16; ++e) {
                const int c = kbase + e + 32 * ch;
                float f = z[zRowBase + (size_t)c * HW];
                unsigned short h = f2bf(f);
                Ahi[g][ch].u[e] = h;
                Alo[g][ch].u[e] = f2bf(f - bf2f(h));
                acc += f * f;
            }
        }
        acc += __shfl_xor(acc, 16, 32);  // combine the two half-rows
        zn[g] = acc;
    }

    __syncthreads();

    // ---- stream all k-tiles: 12 bf16 WMMAs/tile (3xBF16 fp32 emulation,
    //      two independent accumulator chains), argmin update pipelined
    //      one tile behind to avoid WMMA->VALU hazard stalls. ----
    float minv[2][8];
    int   mini[2][8];
#pragma unroll
    for (int g = 0; g < 2; ++g)
#pragma unroll
        for (int r = 0; r < 8; ++r) { minv[g][r] = 3.4e38f; mini[g][r] = 0; }

    const int laneN = lane & 15;
    v8f accP0, accP1;
#pragma unroll
    for (int r = 0; r < 8; ++r) { accP0[r] = -1.0e30f; accP1[r] = -1.0e30f; }
    float enP   = 0.f;   // dummy prev-tile: score = 0 - 2*(-1e30) = +2e30 -> no-op
    int   codeP = 0;

    for (int t = 0; t < nTiles; ++t) {
        const char* bt = sB + (size_t)t * 4096 + (size_t)lane * 32;
        v16bf Bh0 = *(const v16bf*)(bt + 0 * 1024);
        v16bf Bh1 = *(const v16bf*)(bt + 1 * 1024);
        v16bf Bl0 = *(const v16bf*)(bt + 2 * 1024);
        v16bf Bl1 = *(const v16bf*)(bt + 3 * 1024);

        v8f a0 = {}, a1 = {};
        a0 = __builtin_amdgcn_wmma_f32_16x16x32_bf16(false, Ahi[0][0].v, false, Bh0, (short)0, a0, false, false);
        a1 = __builtin_amdgcn_wmma_f32_16x16x32_bf16(false, Ahi[1][0].v, false, Bh0, (short)0, a1, false, false);
        a0 = __builtin_amdgcn_wmma_f32_16x16x32_bf16(false, Ahi[0][1].v, false, Bh1, (short)0, a0, false, false);
        a1 = __builtin_amdgcn_wmma_f32_16x16x32_bf16(false, Ahi[1][1].v, false, Bh1, (short)0, a1, false, false);
        a0 = __builtin_amdgcn_wmma_f32_16x16x32_bf16(false, Ahi[0][0].v, false, Bl0, (short)0, a0, false, false);
        a1 = __builtin_amdgcn_wmma_f32_16x16x32_bf16(false, Ahi[1][0].v, false, Bl0, (short)0, a1, false, false);
        a0 = __builtin_amdgcn_wmma_f32_16x16x32_bf16(false, Ahi[0][1].v, false, Bl1, (short)0, a0, false, false);
        a1 = __builtin_amdgcn_wmma_f32_16x16x32_bf16(false, Ahi[1][1].v, false, Bl1, (short)0, a1, false, false);
        a0 = __builtin_amdgcn_wmma_f32_16x16x32_bf16(false, Alo[0][0].v, false, Bh0, (short)0, a0, false, false);
        a1 = __builtin_amdgcn_wmma_f32_16x16x32_bf16(false, Alo[1][0].v, false, Bh0, (short)0, a1, false, false);
        a0 = __builtin_amdgcn_wmma_f32_16x16x32_bf16(false, Alo[0][1].v, false, Bh1, (short)0, a0, false, false);
        a1 = __builtin_amdgcn_wmma_f32_16x16x32_bf16(false, Alo[1][1].v, false, Bh1, (short)0, a1, false, false);

        // ---- argmin update for the PREVIOUS tile (co-executes under WMMAs) ----
#pragma unroll
        for (int r = 0; r < 8; ++r) {
            float s0 = fmaf(-2.0f, accP0[r], enP);   // ||e||^2 - 2 z.e
            float s1 = fmaf(-2.0f, accP1[r], enP);
            if (s0 < minv[0][r]) { minv[0][r] = s0; mini[0][r] = codeP; }
            if (s1 < minv[1][r]) { minv[1][r] = s1; mini[1][r] = codeP; }
        }

        accP0 = a0;
        accP1 = a1;
        enP   = sNorm[t * 16 + laneN];
        codeP = t * 16 + laneN;
    }
    // ---- drain the pipeline: last tile's update ----
#pragma unroll
    for (int r = 0; r < 8; ++r) {
        float s0 = fmaf(-2.0f, accP0[r], enP);
        float s1 = fmaf(-2.0f, accP1[r], enP);
        if (s0 < minv[0][r]) { minv[0][r] = s0; mini[0][r] = codeP; }
        if (s1 < minv[1][r]) { minv[1][r] = s1; mini[1][r] = codeP; }
    }

    // ---- deposit per-slot results: C layout row M = r + (lane>=16)*8 ----
    const int mrow0 = (lane < 16) ? 0 : 8;
#pragma unroll
    for (int g = 0; g < 2; ++g) {
#pragma unroll
        for (int r = 0; r < 8; ++r) {
            const int rowLocal = wave * 32 + g * 16 + mrow0 + r;
            sMinV[rowLocal * 16 + laneN] = minv[g][r];
            sMinI[rowLocal * 16 + laneN] = mini[g][r];
        }
        if (lane < 16) sZn[wave * 32 + g * 16 + lane] = zn[g];
    }
    __syncthreads();

    // ---- per-row argmin across 16 code-lanes; indices + loss (1 row/thread) ----
    {
        float bv = sMinV[tid * 16];
        int   bi = sMinI[tid * 16];
#pragma unroll
        for (int j = 1; j < 16; ++j) {
            float v = sMinV[tid * 16 + j];
            if (v < bv) { bv = v; bi = sMinI[tid * 16 + j]; }
        }
        sIdx[tid] = bi;
        const int n = blockIdx.x * ROWS_PER_BLOCK + tid;
        out_idx[n] = (float)bi;
        // d_min = s_min + ||z||^2 ; loss = (1+beta)*mean(d_min)
        atomicAdd(sLoss, (bv + sZn[tid]) * lossScale);
    }
    __syncthreads();

    // ---- gather z_q = emb[idx] and scatter transposed [B,C,H,W] ----
    const int n0 = blockIdx.x * ROWS_PER_BLOCK;
#pragma unroll 4
    for (int i = 0; i < (ROWS_PER_BLOCK * C_DIM) / 256; ++i) {
        const int flat = tid + 256 * i;        // row-fast for coalescing
        const int c = flat >> 8;               // / ROWS_PER_BLOCK
        const int r = flat & (ROWS_PER_BLOCK - 1);
        const int id = sIdx[r];
        const float v = emb[id * C_DIM + c];
        const int n = n0 + r;
        out_zq[(size_t)(n >> 12) * C_DIM * HW + (size_t)c * HW + (n & 4095)] = v;
    }
    if (tid == 0) atomicAdd(out_loss, sLoss[0]);
}

extern "C" void kernel_launch(void* const* d_in, const int* in_sizes, int n_in,
                              void* d_out, int out_size, void* d_ws, size_t ws_size,
                              hipStream_t stream) {
    const float* z   = (const float*)d_in[0];
    const float* emb = (const float*)d_in[1];

    const int NC = in_sizes[0];          // B*C*H*W
    const int N  = NC / C_DIM;           // B*H*W rows
    const int K  = in_sizes[1] / C_DIM;  // 1024 codes

    float* out      = (float*)d_out;
    float* out_zq   = out;
    float* out_idx  = out + (size_t)NC;
    float* out_loss = out + (size_t)out_size - 1;
    unsigned char* ws = (unsigned char*)d_ws;

    const size_t shmem = (size_t)(K / 16) * 4096   // bf16 fragments  (256 KB)
                       + (size_t)K * 4             // code norms       (4 KB)
                       + ROWS_PER_BLOCK * 16 * 8   // (minv,mini)     (32 KB)
                       + ROWS_PER_BLOCK * 8        // zn + idx         (2 KB)
                       + 16;                       // loss
    hipFuncSetAttribute((const void*)vq_main,
                        hipFuncAttributeMaxDynamicSharedMemorySize, (int)shmem);

    vq_zero_loss<<<1, 1, 0, stream>>>(out_loss);
    vq_prep<<<K / 16, 32, 0, stream>>>(emb, ws, K);

    const float lossScale = 1.25f / (float)NC;     // (1+BETA)/(N*C)
    vq_main<<<N / ROWS_PER_BLOCK, 256, shmem, stream>>>(
        z, emb, ws, out_zq, out_idx, out_loss, K, lossScale);
}
